// HeteroBidderGNN_64828236366584
// MI455X (gfx1250) — matrix operand order repaired and checked by reading, not compiled, not run on hardware
//
#include <hip/hip_runtime.h>
#include <hip/hip_bf16.h>
#include <math.h>

// ---------------------------------------------------------------------------
// HeteroBidderGNN on MI455X (gfx1250, wave32, WMMA).
// f16 activations + f16 pre-converted weights everywhere; f32 accumulation
// (scatter atomics + multi-GEMM sums). Edge gather/scatter is the HBM-bound
// part (~7 GB total at f16 -> ~0.3 ms @ 23.3 TB/s); GEMMs (~94 GFLOP) run on
// v_wmma_f32_16x16x32_f16 with b128 A-fragment loads (no in-loop cvt).
// ---------------------------------------------------------------------------

#define NB 200000
#define NA 50000
#define ND 30000
#define NI 100000
#define EE 1000000
#define HID 128
#define INDIM 64
#define NSEL 4096

typedef __attribute__((ext_vector_type(16))) _Float16 v16h;
typedef __attribute__((ext_vector_type(8)))  _Float16 v8h;
typedef __attribute__((ext_vector_type(4)))  _Float16 v4h;
typedef __attribute__((ext_vector_type(8)))  float    v8f;

#define GF_ZERO 1    // initialize C tile to zero (else accumulate from f32 C)
#define GF_BIAS 2    // add bias[col] on store
#define GF_IDX  4    // A rows gathered through rowidx[]
#define GF_RELU 8    // relu on store
#define GF_F16  16   // final store to f16 buffer Ch (f32 C not written)
#define GF_GELU 32   // exact gelu on store

// ---------------------------------------------------------------------------
// WMMA GEMM: Cacc[M,N] (+)= A[M,K] @ B[K,N] (+bias, +relu/gelu, f16-out).
// A: f16 row-major stride K. B: f16 row-major stride N. C: f32 accumulator.
// One wave computes a 16 x N strip (NT = N/16 tiles), A fragment reused
// across all N tiles. M%16==0, K%32==0, N==NT*16. Wave32; EXEC all-ones.
// Fragment layouts per ISA 7.12.2 (16-bit A 16x32; B lane<->N, elem<->K;
// C/D 8 VGPRs with lanes 16-31 holding M+8).
// ---------------------------------------------------------------------------
template <int NT>
__global__ void wmma_gemm_kernel(const _Float16* __restrict__ A,
                                 const _Float16* __restrict__ B,
                                 const float* __restrict__ bias,
                                 float* __restrict__ C,
                                 _Float16* __restrict__ Ch,
                                 const int* __restrict__ rowidx,
                                 int M, int N, int K, int flags)
{
    const int lane  = threadIdx.x & 31;
    const int wave  = (blockIdx.x * blockDim.x + threadIdx.x) >> 5;
    const int row0  = wave * 16;
    if (row0 >= M) return;                 // uniform per wave -> EXEC stays full

    const int mloc = lane & 15;
    const int hi   = lane >> 4;
    int arow = row0 + mloc;
    if (flags & GF_IDX) arow = rowidx[arow];
    const _Float16* __restrict__ Arow = A + (size_t)arow * (size_t)K;

    v8f acc[NT];
    if (flags & GF_ZERO) {
        #pragma unroll
        for (int nt = 0; nt < NT; ++nt)
            #pragma unroll
            for (int r = 0; r < 8; ++r) acc[nt][r] = 0.0f;
    } else {
        #pragma unroll
        for (int nt = 0; nt < NT; ++nt)
            #pragma unroll
            for (int r = 0; r < 8; ++r)
                acc[nt][r] = C[(size_t)(row0 + r + 8 * hi) * N + nt * 16 + mloc];
    }

    for (int k0 = 0; k0 < K; k0 += 32) {
        // A fragment 16x32: hi=0 lanes hold K=k0+[0..7],[16..23];
        // hi=1 lanes hold K=k0+[8..15],[24..31]. Two 16B loads, no cvt.
        const v8h alo = *(const v8h*)(Arow + k0 + hi * 8);
        const v8h ahi = *(const v8h*)(Arow + k0 + 16 + hi * 8);
        v16h a;
        #pragma unroll
        for (int j = 0; j < 8; ++j) { a[j] = alo[j]; a[8 + j] = ahi[j]; }

        #pragma unroll
        for (int nt = 0; nt < NT; ++nt) {
            // B fragment 32x16: lane <-> N col, elem <-> K row (same K order as A).
            v16h b;
            #pragma unroll
            for (int j = 0; j < 8; ++j) {
                b[j]     = B[(size_t)(k0 + hi * 8 + j) * N + nt * 16 + mloc];
                b[8 + j] = B[(size_t)(k0 + 16 + hi * 8 + j) * N + nt * 16 + mloc];
            }
            acc[nt] = __builtin_amdgcn_wmma_f32_16x16x32_f16(
                false, a, false, b, (short)0, acc[nt], false, false);
        }
    }

    #pragma unroll
    for (int nt = 0; nt < NT; ++nt) {
        #pragma unroll
        for (int r = 0; r < 8; ++r) {
            const int col = nt * 16 + mloc;
            const size_t idx = (size_t)(row0 + r + 8 * hi) * N + col;
            float v = acc[nt][r];
            if (flags & GF_BIAS) v += bias[col];
            if (flags & GF_RELU) v = fmaxf(v, 0.0f);
            if (flags & GF_GELU) v = 0.5f * v * (1.0f + erff(v * 0.70710678118654752f));
            if (flags & GF_F16)  Ch[idx] = (_Float16)v;
            else                 C[idx]  = v;
        }
    }
}

// ---------------------------------------------------------------------------
// LayerNorm over rows of 64; one wave per row; emits f16 directly.
// ---------------------------------------------------------------------------
__global__ void ln_kernel(const float* __restrict__ x, const float* __restrict__ g,
                          const float* __restrict__ b, _Float16* __restrict__ y, int M)
{
    const int wave = (blockIdx.x * blockDim.x + threadIdx.x) >> 5;
    const int lane = threadIdx.x & 31;
    if (wave >= M) return;
    const float2 v = ((const float2*)(x + (size_t)wave * INDIM))[lane];
    float s = v.x + v.y;
    #pragma unroll
    for (int o = 16; o > 0; o >>= 1) s += __shfl_xor(s, o, 32);
    const float mean = s * (1.0f / 64.0f);
    const float dx = v.x - mean, dy = v.y - mean;
    float q = dx * dx + dy * dy;
    #pragma unroll
    for (int o = 16; o > 0; o >>= 1) q += __shfl_xor(q, o, 32);
    const float rstd = rsqrtf(q * (1.0f / 64.0f) + 1e-5f);
    const float2 gg = ((const float2*)g)[lane];
    const float2 bb = ((const float2*)b)[lane];
    v4h dummy; (void)dummy;
    _Float16 o0 = (_Float16)(dx * rstd * gg.x + bb.x);
    _Float16 o1 = (_Float16)(dy * rstd * gg.y + bb.y);
    _Float16* orow = y + (size_t)wave * INDIM + lane * 2;
    orow[0] = o0; orow[1] = o1;
}

// ---------------------------------------------------------------------------
// Edge scatter: one wave per edge; 32 lanes x 4 f16 = 128 features.
// 256B coalesced f16 row read, f32 native atomic adds into agg[dst].
// ---------------------------------------------------------------------------
__global__ void scatter_kernel(const _Float16* __restrict__ xsrc,
                               const int* __restrict__ src,
                               const int* __restrict__ dst,
                               float* __restrict__ agg,
                               float* __restrict__ deg, int nedge)
{
    const int wave = (blockIdx.x * blockDim.x + threadIdx.x) >> 5;
    const int lane = threadIdx.x & 31;
    if (wave >= nedge) return;
    const int s = src[wave];
    const int d = dst[wave];
    const v4h v = *(const v4h*)(xsrc + (size_t)s * HID + lane * 4);
    float* out = agg + (size_t)d * HID + lane * 4;
    __hip_atomic_fetch_add(out + 0, (float)v[0], __ATOMIC_RELAXED, __HIP_MEMORY_SCOPE_AGENT);
    __hip_atomic_fetch_add(out + 1, (float)v[1], __ATOMIC_RELAXED, __HIP_MEMORY_SCOPE_AGENT);
    __hip_atomic_fetch_add(out + 2, (float)v[2], __ATOMIC_RELAXED, __HIP_MEMORY_SCOPE_AGENT);
    __hip_atomic_fetch_add(out + 3, (float)v[3], __ATOMIC_RELAXED, __HIP_MEMORY_SCOPE_AGENT);
    if (lane == 0)
        __hip_atomic_fetch_add(deg + d, 1.0f, __ATOMIC_RELAXED, __HIP_MEMORY_SCOPE_AGENT);
}

__global__ void zero_kernel(float* __restrict__ p, long long n)
{
    long long i = (long long)blockIdx.x * blockDim.x + threadIdx.x;
    if (i < n) p[i] = 0.0f;
}

// mean = agg / max(deg,1), emitted as f16 (consumed only by WMMA GEMM A-side)
__global__ void mean_kernel(const float* __restrict__ agg, const float* __restrict__ deg,
                            _Float16* __restrict__ mh, long long n)
{
    long long i = (long long)blockIdx.x * blockDim.x + threadIdx.x;
    if (i < n) {
        float d = deg[i >> 7];
        mh[i] = (_Float16)(agg[i] / fmaxf(d, 1.0f));
    }
}

// f32 -> f16 copy (weight pre-conversion)
__global__ void cvt_kernel(const float* __restrict__ s, _Float16* __restrict__ d,
                           long long n)
{
    long long i = (long long)blockIdx.x * blockDim.x + threadIdx.x;
    if (i < n) d[i] = (_Float16)s[i];
}

// Final head: out[i] = dot(z[i, :64], W2) + b2
__global__ void head2_kernel(const float* __restrict__ z, const float* __restrict__ W2,
                             const float* __restrict__ b2, float* __restrict__ out, int M)
{
    int i = blockIdx.x * blockDim.x + threadIdx.x;
    if (i >= M) return;
    const float* r = z + (size_t)i * 64;
    float acc = 0.0f;
    #pragma unroll
    for (int j = 0; j < 64; ++j) acc += r[j] * W2[j];
    out[i] = acc + b2[0];
}

// ---------------------------------------------------------------------------
// Host-side orchestration
// ---------------------------------------------------------------------------
static inline void launch_gemm(const _Float16* A, const _Float16* B, const float* bias,
                               float* C, _Float16* Ch, const int* idx,
                               int M, int N, int K, int flags, hipStream_t s)
{
    int waves = M / 16;
    int wpb = 4;                        // 128 threads / block
    int blocks = (waves + wpb - 1) / wpb;
    dim3 grid(blocks), block(wpb * 32);
    if (N == 128)
        wmma_gemm_kernel<8><<<grid, block, 0, s>>>(A, B, bias, C, Ch, idx, M, N, K, flags);
    else
        wmma_gemm_kernel<4><<<grid, block, 0, s>>>(A, B, bias, C, Ch, idx, M, N, K, flags);
}

static inline void launch_zero(float* p, long long n, hipStream_t s)
{
    zero_kernel<<<(unsigned)((n + 255) / 256), 256, 0, s>>>(p, n);
}

extern "C" void kernel_launch(void* const* d_in, const int* in_sizes, int n_in,
                              void* d_out, int out_size, void* d_ws, size_t ws_size,
                              hipStream_t stream)
{
    (void)in_sizes; (void)n_in; (void)out_size; (void)ws_size;

    // ---- inputs (setup_inputs dict order; params flattened in insertion order)
    const float* bidder_x = (const float*)d_in[0];
    const int*   edges[6];                       // b2a,a2b,b2d,d2b,b2i,i2b : [2,E]
    for (int i = 0; i < 6; ++i) edges[i] = (const int*)d_in[1 + i];
    const int*   node_idx = (const int*)d_in[7];

    const float* ln_gamma    = (const float*)d_in[8];
    const float* ln_beta     = (const float*)d_in[9];
    const float* proj_W      = (const float*)d_in[10];
    const float* proj_b      = (const float*)d_in[11];
    const float* auction_emb = (const float*)d_in[12];
    const float* device_emb  = (const float*)d_in[13];
    const float* ip_emb      = (const float*)d_in[14];
    // conv1: d_in[15..32] = {Wl,bl,Wr} x {b2a,a2b,b2d,d2b,b2i,i2b}
    // conv2: d_in[33..50]
    const float* head_W1 = (const float*)d_in[51];
    const float* head_b1 = (const float*)d_in[52];
    const float* head_W2 = (const float*)d_in[53];
    const float* head_b2 = (const float*)d_in[54];

    float* out = (float*)d_out;

    // ---- workspace: f32 region (accumulators) then f16 region (~686 MB)
    float* f = (float*)d_ws;
    size_t o = 0;
    float* h1acc[4];
    h1acc[0] = f + o; o += (size_t)NB * HID;       // bidder
    h1acc[1] = f + o; o += (size_t)NA * HID;       // auction
    h1acc[2] = f + o; o += (size_t)ND * HID;       // device
    h1acc[3] = f + o; o += (size_t)NI * HID;       // ip
    float* h2acc  = f + o; o += (size_t)NB * HID;
    float* agg    = f + o; o += (size_t)NB * HID;  // max dst, reused per edge type
    float* deg    = f + o; o += (size_t)NB;
    float* head_z = f + o; o += (size_t)NSEL * 64;

    _Float16* h = (_Float16*)(f + o);
    size_t ho = 0;
    _Float16* ln_h   = h + ho; ho += (size_t)NB * INDIM;
    _Float16* x_h    = h + ho; ho += (size_t)NB * HID;     // bidder f16 features
    _Float16* mean_h = h + ho; ho += (size_t)NB * HID;     // reused per edge type
    _Float16* h1h[4];
    h1h[0] = h + ho; ho += (size_t)NB * HID;
    h1h[1] = h + ho; ho += (size_t)NA * HID;
    h1h[2] = h + ho; ho += (size_t)ND * HID;
    h1h[3] = h + ho; ho += (size_t)NI * HID;
    _Float16* h2h = h + ho; ho += (size_t)NB * HID;
    // f16 weights
    _Float16* proj_Wh = h + ho; ho += (size_t)INDIM * HID;
    _Float16* c1Wl[6], *c1Wr[6], *c2Wl[6], *c2Wr[6];
    for (int e = 0; e < 6; ++e) { c1Wl[e] = h + ho; ho += HID * HID;
                                  c1Wr[e] = h + ho; ho += HID * HID; }
    const int elist2[3] = {1, 3, 5};
    for (int k = 0; k < 6; ++k) { c2Wl[k] = nullptr; c2Wr[k] = nullptr; }
    for (int k = 0; k < 3; ++k) { int e = elist2[k];
                                  c2Wl[e] = h + ho; ho += HID * HID;
                                  c2Wr[e] = h + ho; ho += HID * HID; }
    _Float16* head_W1h = h + ho; ho += (size_t)HID * 64;

    // ---- 0) pre-convert weights to f16 (tiny, ~620 KB total)
    {
        auto cvt = [&](const float* s, _Float16* d, long long n) {
            cvt_kernel<<<(unsigned)((n + 255) / 256), 256, 0, stream>>>(s, d, n);
        };
        cvt(proj_W, proj_Wh, (long long)INDIM * HID);
        for (int e = 0; e < 6; ++e) {
            cvt((const float*)d_in[15 + 3 * e + 0], c1Wl[e], HID * HID);
            cvt((const float*)d_in[15 + 3 * e + 2], c1Wr[e], HID * HID);
        }
        for (int k = 0; k < 3; ++k) {
            int e = elist2[k];
            cvt((const float*)d_in[33 + 3 * e + 0], c2Wl[e], HID * HID);
            cvt((const float*)d_in[33 + 3 * e + 2], c2Wr[e], HID * HID);
        }
        cvt(head_W1, head_W1h, (long long)HID * 64);
    }

    // embeddings (f32 inputs) -> f16 feature tables for auction/device/ip
    _Float16* emb_h[4] = {nullptr, nullptr, nullptr, nullptr};
    {
        // reuse tails: convert into h1h buffers? No: h1h needed later. Allocate:
        emb_h[1] = h + ho; ho += (size_t)NA * HID;
        emb_h[2] = h + ho; ho += (size_t)ND * HID;
        emb_h[3] = h + ho; ho += (size_t)NI * HID;
        cvt_kernel<<<(unsigned)(((long long)NA * HID + 255) / 256), 256, 0, stream>>>(
            auction_emb, emb_h[1], (long long)NA * HID);
        cvt_kernel<<<(unsigned)(((long long)ND * HID + 255) / 256), 256, 0, stream>>>(
            device_emb, emb_h[2], (long long)ND * HID);
        cvt_kernel<<<(unsigned)(((long long)NI * HID + 255) / 256), 256, 0, stream>>>(
            ip_emb, emb_h[3], (long long)NI * HID);
    }

    // edge metadata: 0=bidder,1=auction,2=device,3=ip
    const int e_src[6] = {0, 1, 0, 2, 0, 3};
    const int e_dst[6] = {1, 0, 2, 0, 3, 0};
    const int n_of[4]  = {NB, NA, ND, NI};

    // ---- 1) LayerNorm (f16 out) + projection -> x_h [NB,128] f16
    ln_kernel<<<(NB * 32 + 255) / 256, 256, 0, stream>>>(
        bidder_x, ln_gamma, ln_beta, ln_h, NB);
    launch_gemm(ln_h, proj_Wh, proj_b, nullptr, x_h, nullptr,
                NB, HID, INDIM, GF_ZERO | GF_BIAS | GF_F16, stream);

    const _Float16* x1[4] = {x_h, emb_h[1], emb_h[2], emb_h[3]};
    const int scatter_blocks = (EE * 32) / 256;   // one wave per edge

    // ---- 2) conv1: all 6 edge types, f32-accumulate into h1acc[dt];
    //         last GEMM per dst fuses relu + f16 store into h1h[dt].
    {
        bool seen[4] = {false, false, false, false};
        for (int e = 0; e < 6; ++e) {
            const int st = e_src[e], dt = e_dst[e];
            const int ndst = n_of[dt];
            const long long nagg = (long long)ndst * HID;
            launch_zero(agg, nagg, stream);
            launch_zero(deg, ndst, stream);
            scatter_kernel<<<scatter_blocks, 256, 0, stream>>>(
                x1[st], edges[e], edges[e] + EE, agg, deg, EE);
            mean_kernel<<<(unsigned)((nagg + 255) / 256), 256, 0, stream>>>(
                agg, deg, mean_h, nagg);

            const float* bl = (const float*)d_in[15 + 3 * e + 1];
            const bool last = (dt != 0) || (e == 5);   // bidder finishes at i2b
            int f0 = (seen[dt] ? 0 : GF_ZERO) | GF_BIAS;
            launch_gemm(mean_h, c1Wl[e], bl, h1acc[dt], nullptr, nullptr,
                        ndst, HID, HID, f0, stream);
            int f1 = last ? (GF_RELU | GF_F16) : 0;
            launch_gemm(x1[dt], c1Wr[e], nullptr, h1acc[dt], h1h[dt], nullptr,
                        ndst, HID, HID, f1, stream);
            seen[dt] = true;
        }
    }

    // ---- 3) conv2: only dst == bidder (a2b=1, d2b=3, i2b=5) -> h2h
    {
        bool first = true;
        for (int k = 0; k < 3; ++k) {
            const int e = elist2[k];
            const int st = e_src[e];
            const long long nagg = (long long)NB * HID;
            launch_zero(agg, nagg, stream);
            launch_zero(deg, NB, stream);
            scatter_kernel<<<scatter_blocks, 256, 0, stream>>>(
                h1h[st], edges[e], edges[e] + EE, agg, deg, EE);
            mean_kernel<<<(unsigned)((nagg + 255) / 256), 256, 0, stream>>>(
                agg, deg, mean_h, nagg);

            const float* bl = (const float*)d_in[33 + 3 * e + 1];
            int f0 = (first ? GF_ZERO : 0) | GF_BIAS;
            launch_gemm(mean_h, c2Wl[e], bl, h2acc, nullptr, nullptr,
                        NB, HID, HID, f0, stream);
            int f1 = (e == 5) ? (GF_RELU | GF_F16) : 0;
            launch_gemm(h1h[0], c2Wr[e], nullptr, h2acc, h2h, nullptr,
                        NB, HID, HID, f1, stream);
            first = false;
        }
    }

    // ---- 4) head: gather(node_idx) fused into A-load; GELU fused into store
    launch_gemm(h2h, head_W1h, head_b1, head_z, nullptr, node_idx,
                NSEL, 64, HID, GF_ZERO | GF_BIAS | GF_IDX | GF_GELU, stream);
    head2_kernel<<<(NSEL + 255) / 256, 256, 0, stream>>>(head_z, head_W2, head_b2, out, NSEL);
}
